// CausalSelfAttention_3582002725485
// MI455X (gfx1250) — matrix operand order repaired and checked by reading, not compiled, not run on hardware
//
#include <hip/hip_runtime.h>

typedef __attribute__((ext_vector_type(16))) __bf16 bf16x16;
typedef __attribute__((ext_vector_type(8)))  __bf16 bf16x8;
typedef __attribute__((ext_vector_type(8)))  float  f32x8;

union BF16Frag { bf16x16 v; bf16x8 h[2]; };

static __device__ __forceinline__ f32x8 wmma_bf16f32(bf16x16 a, bf16x16 b, f32x8 c) {
  // D = A(16x32 bf16) x B(32x16 bf16) + C(16x16 f32)
  return __builtin_amdgcn_wmma_f32_16x16x32_bf16(false, a, false, b, (short)0, c, false, false);
}

// Problem constants
#define CB 4
#define CT 2048
#define CD 1024
#define CH 16
#define CHD 64

// ---------------------------------------------------------------------------
// Async global->LDS copy (CDNA5 GLOBAL_LOAD_ASYNC_TO_LDS_B128, ASYNCcnt),
// with a safe fallback if the builtin is not available in this toolchain.
// Probe-derived signature: (int4 AS1*, int4 AS3*, imm offset, imm cpol).
// ---------------------------------------------------------------------------
#if __has_builtin(__builtin_amdgcn_global_load_async_to_lds_b128)
#define HAVE_ASYNC_LDS 1
typedef int gl_v4i __attribute__((vector_size(4 * sizeof(int))));
typedef __attribute__((address_space(1))) gl_v4i* gl_v4i_as1p;
typedef __attribute__((address_space(3))) gl_v4i* gl_v4i_as3p;
#endif

static __device__ __forceinline__ void async_copy16(const __bf16* g, __bf16* l) {
#ifdef HAVE_ASYNC_LDS
  __builtin_amdgcn_global_load_async_to_lds_b128(
      (gl_v4i_as1p)(uintptr_t)g, (gl_v4i_as3p)(uintptr_t)l, 0, 0);
#else
  *(bf16x8*)l = *(const bf16x8*)g;
#endif
}

static __device__ __forceinline__ void wait_async() {
#ifdef HAVE_ASYNC_LDS
#if __has_builtin(__builtin_amdgcn_s_wait_asynccnt)
  __builtin_amdgcn_s_wait_asynccnt(0);
#else
  asm volatile("s_wait_asynccnt 0x0" ::: "memory");
#endif
#endif
}

// ---------------------------------------------------------------------------
// GEMM tile stagers (f32 -> bf16 conversion happens in VGPRs).
// LDS layouts: A [row][k] stride 40 (A-fragment: two 8-elem K chunks / lane),
//              B [n][k]  stride 48 (B-fragment: two 8-elem K chunks / lane).
// ---------------------------------------------------------------------------
#define GLDA 40
#define GLDB 48

static __device__ __forceinline__ void g2s_a_f32(
    const float* __restrict__ x, __bf16* sA, int m0, int k0, int Kd, int tid) {
  #pragma unroll
  for (int i = 0; i < 4; i++) {
    int idx = tid + i * 256;            // 0..1023 : 128 rows x 8 float4
    int row = idx >> 3, kq = (idx & 7) * 4;
    float4 dv = *(const float4*)(x + (size_t)(m0 + row) * Kd + k0 + kq);
    __bf16* d = &sA[row * GLDA + kq];
    d[0] = (__bf16)dv.x; d[1] = (__bf16)dv.y;
    d[2] = (__bf16)dv.z; d[3] = (__bf16)dv.w;
  }
}

static __device__ __forceinline__ void g2s_a_bf16(
    const __bf16* __restrict__ a, __bf16* sA, int m0, int k0, int Kd, int tid) {
  #pragma unroll
  for (int i = 0; i < 2; i++) {
    int idx = tid + i * 256;            // 0..511 : 128 rows x 4 chunks
    int row = idx >> 2, kq = (idx & 3) * 8;
    *(bf16x8*)&sA[row * GLDA + kq] =
        *(const bf16x8*)(a + (size_t)(m0 + row) * Kd + k0 + kq);
  }
}

static __device__ __forceinline__ void g2s_b_f32(
    const float* __restrict__ w, __bf16* sB, int n0, int k0, int N, int tid) {
  #pragma unroll
  for (int i = 0; i < 2; i++) {
    int idx = tid + i * 256;            // 0..511 : 32 k-rows x 16 float4
    int kk = idx >> 4, nq = (idx & 15) * 4;
    float4 dv = *(const float4*)(w + (size_t)(k0 + kk) * N + n0 + nq);
    sB[(nq + 0) * GLDB + kk] = (__bf16)dv.x;
    sB[(nq + 1) * GLDB + kk] = (__bf16)dv.y;
    sB[(nq + 2) * GLDB + kk] = (__bf16)dv.z;
    sB[(nq + 3) * GLDB + kk] = (__bf16)dv.w;
  }
}

// Wave-level 32x32 MMA step from staged tiles (4 WMMA).
static __device__ __forceinline__ void mma_step(
    const __bf16* sA, const __bf16* sB, int wm0, int wn0, int n16, int half,
    f32x8 acc[2][2]) {
  BF16Frag a[2]; BF16Frag bf[2];
  #pragma unroll
  for (int mt = 0; mt < 2; mt++) {
    const __bf16* p = &sA[(wm0 + mt * 16 + n16) * GLDA];
    a[mt].h[0] = *(const bf16x8*)(p + half * 8);        // K 0..7 / 8..15
    a[mt].h[1] = *(const bf16x8*)(p + 16 + half * 8);   // K 16..23 / 24..31
  }
  #pragma unroll
  for (int nt = 0; nt < 2; nt++) {
    bf[nt].h[0] = *(const bf16x8*)&sB[(wn0 + nt * 16 + n16) * GLDB + half * 8];
    bf[nt].h[1] = *(const bf16x8*)&sB[(wn0 + nt * 16 + n16) * GLDB + 16 + half * 8];
  }
  #pragma unroll
  for (int mt = 0; mt < 2; mt++)
    #pragma unroll
    for (int nt = 0; nt < 2; nt++)
      acc[mt][nt] = wmma_bf16f32(a[mt].v, bf[nt].v, acc[mt][nt]);
}

// ---------------------------------------------------------------------------
// Kernel 1: qkv = x @ Wqkv + bqkv ; scatter into Q/K bf16 [B,H,T,HD],
//           V pre-TRANSPOSED as [B,H,HD,T] (B-operand layout for P·V).
// M = 8192, N = 3072, K = 1024. Double-buffered LDS, 1 barrier / K-step.
// ---------------------------------------------------------------------------
__global__ __launch_bounds__(256) void qkv_gemm_kernel(
    const float* __restrict__ x, const float* __restrict__ w,
    const float* __restrict__ bias,
    __bf16* __restrict__ Q, __bf16* __restrict__ K, __bf16* __restrict__ V)
{
  constexpr int Kd = CD, N = 3 * CD, NSTEP = Kd / 32;
  __shared__ __align__(32) __bf16 sA[2 * 128 * GLDA];
  __shared__ __align__(32) __bf16 sB[2 * 64 * GLDB];

  const int tid  = threadIdx.x;
  const int m0   = blockIdx.x * 128;
  const int n0   = blockIdx.y * 64;
  const int wave = tid >> 5, lane = tid & 31;
  const int n16  = lane & 15, half = lane >> 4;
  const int wm0  = (wave & 3) * 32;
  const int wn0  = (wave >> 2) * 32;

  f32x8 acc[2][2] = {};

  g2s_a_f32(x, sA, m0, 0, Kd, tid);
  g2s_b_f32(w, sB, n0, 0, N, tid);
  __syncthreads();

  for (int kt = 0; kt < NSTEP; kt++) {
    if (kt + 1 < NSTEP) {
      int nb = (kt + 1) & 1;
      g2s_a_f32(x, sA + nb * 128 * GLDA, m0, (kt + 1) * 32, Kd, tid);
      g2s_b_f32(w, sB + nb * 64 * GLDB, n0, (kt + 1) * 32, N, tid);
    }
    mma_step(sA + (kt & 1) * 128 * GLDA, sB + (kt & 1) * 64 * GLDB,
             wm0, wn0, n16, half, acc);
    __syncthreads();
  }

  // Epilogue: bias + scatter
  #pragma unroll
  for (int mt = 0; mt < 2; mt++) {
    #pragma unroll
    for (int nt = 0; nt < 2; nt++) {
      int gn = n0 + wn0 + nt * 16 + n16;
      float bs = bias[gn];
      int which = gn >> 10, rem = gn & 1023;
      int hh = rem >> 6, d = rem & 63;
      #pragma unroll
      for (int r = 0; r < 8; r++) {
        int gm = m0 + wm0 + mt * 16 + 8 * half + r;
        int bb = gm >> 11, tt = gm & (CT - 1);
        __bf16 val = (__bf16)(acc[mt][nt][r] + bs);
        if (which == 2) {
          V[((size_t)(bb * CH + hh) * CHD + d) * CT + tt] = val;   // V^T
        } else {
          __bf16* dst = which ? K : Q;
          dst[((size_t)(bb * CH + hh) * CT + tt) * CHD + d] = val;
        }
      }
    }
  }
}

// ---------------------------------------------------------------------------
// Kernel 2: causal flash attention. Block = 128 q rows of one (b,h),
// 8 waves x 16 rows. Per 64-key tile: 8 WMMA (Q·K^T) + 8 WMMA (P·V).
// K tile [key][hd], V^T tile [hd][key] async-DMA'd into double-buffered LDS.
// ---------------------------------------------------------------------------
#define ALDK 80
#define ALDP 72
#define KVTILE (64 * ALDK)

static __device__ __forceinline__ void attn_copy_tiles(
    const __bf16* __restrict__ Kg, const __bf16* __restrict__ Vtg,
    __bf16* sKbuf, __bf16* sVbuf, int kbase, int tid)
{
  #pragma unroll
  for (int i = 0; i < 2; i++) {
    int idx = tid + i * 256;            // 0..511 : 64 rows x 8 chunks
    int row = idx >> 3, c = (idx & 7) * 8;
    async_copy16(Kg + (size_t)(kbase + row) * CHD + c, &sKbuf[row * ALDK + c]);
    async_copy16(Vtg + (size_t)row * CT + kbase + c, &sVbuf[row * ALDK + c]);
  }
}

__global__ __launch_bounds__(256) void attn_fwd_kernel(
    const __bf16* __restrict__ Qm, const __bf16* __restrict__ Km,
    const __bf16* __restrict__ Vt, __bf16* __restrict__ O)
{
  __shared__ __align__(32) __bf16 sK [2 * KVTILE];   // [key][hd]
  __shared__ __align__(32) __bf16 sVt[2 * KVTILE];   // [hd][key]
  __shared__ __align__(32) __bf16 sP [8 * 16 * ALDP];

  const int tid  = threadIdx.x;
  const int wave = tid >> 5, lane = tid & 31;
  const int n16  = lane & 15, half = lane >> 4;
  const int qt = blockIdx.x, bh = blockIdx.y;
  const int q0 = qt * 128 + wave * 16;
  const __bf16* Kg  = Km + (size_t)bh * CT * CHD;
  const __bf16* Vtg = Vt + (size_t)bh * CHD * CT;

  // Q A-fragments, pre-scaled by 1/sqrt(HD)
  BF16Frag aq[2];
  {
    const __bf16* qp = Qm + (size_t)bh * CT * CHD + (size_t)(q0 + n16) * CHD;
    #pragma unroll
    for (int kk = 0; kk < 2; kk++) {
      bf16x8 lo = *(const bf16x8*)(qp + kk * 32 + half * 8);
      bf16x8 hi = *(const bf16x8*)(qp + kk * 32 + 16 + half * 8);
      #pragma unroll
      for (int i = 0; i < 8; i++) {
        aq[kk].v[i]     = (__bf16)((float)lo[i] * 0.125f);
        aq[kk].v[i + 8] = (__bf16)((float)hi[i] * 0.125f);
      }
    }
  }

  float m_r[8], l_r[8];
  #pragma unroll
  for (int r = 0; r < 8; r++) { m_r[r] = -__builtin_inff(); l_r[r] = 0.f; }
  f32x8 acc_o[4] = {};

  __bf16* pw = &sP[wave * 16 * ALDP];
  const int ktmax = 2 * qt + 1;

  attn_copy_tiles(Kg, Vtg, sK, sVt, 0, tid);
  wait_async();
  __syncthreads();

  for (int kt = 0; kt <= ktmax; ++kt) {
    const int kbase = kt * 64;
    if (kt < ktmax) {
      int nb = (kt + 1) & 1;
      attn_copy_tiles(Kg, Vtg, sK + nb * KVTILE, sVt + nb * KVTILE,
                      (kt + 1) * 64, tid);
    }
    const __bf16* sKc  = sK  + (kt & 1) * KVTILE;
    const __bf16* sVtc = sVt + (kt & 1) * KVTILE;

    if (kbase <= q0 + 15) {            // wave-uniform: EXEC all-ones for WMMA
      // S = Q K^T  (16 x 64)
      f32x8 s[4];
      #pragma unroll
      for (int nt = 0; nt < 4; nt++) {
        f32x8 z = {};
        s[nt] = z;
        #pragma unroll
        for (int kk = 0; kk < 2; kk++) {
          BF16Frag bk;
          bk.h[0] = *(const bf16x8*)&sKc[(nt * 16 + n16) * ALDK + kk * 32 + half * 8];
          bk.h[1] = *(const bf16x8*)&sKc[(nt * 16 + n16) * ALDK + kk * 32 + 16 + half * 8];
          s[nt] = wmma_bf16f32(aq[kk].v, bk.v, s[nt]);
        }
      }
      // Causal mask (only when tile crosses the diagonal)
      if (kbase + 63 > q0) {
        #pragma unroll
        for (int nt = 0; nt < 4; nt++) {
          int key = kbase + nt * 16 + n16;
          #pragma unroll
          for (int r = 0; r < 8; r++) {
            int qi = q0 + 8 * half + r;
            if (key > qi) s[nt][r] = -3.0e38f;
          }
        }
      }
      // Online softmax: a row lives in one VGPR across a 16-lane half-wave
      #pragma unroll
      for (int r = 0; r < 8; r++) {
        float mloc = fmaxf(fmaxf(s[0][r], s[1][r]), fmaxf(s[2][r], s[3][r]));
        #pragma unroll
        for (int off = 8; off >= 1; off >>= 1)
          mloc = fmaxf(mloc, __shfl_xor(mloc, off, 32));
        float mnew  = fmaxf(m_r[r], mloc);
        float alpha = __expf(m_r[r] - mnew);
        m_r[r] = mnew;
        float rs = 0.f;
        #pragma unroll
        for (int nt = 0; nt < 4; nt++) {
          float p = __expf(s[nt][r] - mnew);
          s[nt][r] = p;
          rs += p;
        }
        #pragma unroll
        for (int off = 8; off >= 1; off >>= 1)
          rs += __shfl_xor(rs, off, 32);
        l_r[r] = l_r[r] * alpha + rs;
        #pragma unroll
        for (int nt = 0; nt < 4; nt++) acc_o[nt][r] *= alpha;
      }
      // P: C-layout -> A-layout via per-wave LDS staging
      #pragma unroll
      for (int nt = 0; nt < 4; nt++)
        #pragma unroll
        for (int r = 0; r < 8; r++)
          pw[(8 * half + r) * ALDP + nt * 16 + n16] = (__bf16)s[nt][r];
      asm volatile("s_wait_dscnt 0x0" ::: "memory");
      BF16Frag ap[2];
      #pragma unroll
      for (int kk = 0; kk < 2; kk++) {
        ap[kk].h[0] = *(const bf16x8*)&pw[n16 * ALDP + kk * 32 + half * 8];
        ap[kk].h[1] = *(const bf16x8*)&pw[n16 * ALDP + kk * 32 + 16 + half * 8];
      }
      // O += P V
      #pragma unroll
      for (int nt = 0; nt < 4; nt++)
        #pragma unroll
        for (int kk = 0; kk < 2; kk++) {
          BF16Frag bv;
          bv.h[0] = *(const bf16x8*)&sVtc[(nt * 16 + n16) * ALDK + kk * 32 + half * 8];
          bv.h[1] = *(const bf16x8*)&sVtc[(nt * 16 + n16) * ALDK + kk * 32 + 16 + half * 8];
          acc_o[nt] = wmma_bf16f32(ap[kk].v, bv.v, acc_o[nt]);
        }
    }
    wait_async();
    __syncthreads();
  }

  // Normalize and store O as bf16 [B,T,H,HD]
  const int b = bh >> 4, h = bh & 15;
  #pragma unroll
  for (int r = 0; r < 8; r++) {
    float inv = 1.0f / l_r[r];
    int qi = q0 + 8 * half + r;
    size_t rowbase = ((size_t)(b * CT + qi) * CH + h) * CHD;
    #pragma unroll
    for (int nt = 0; nt < 4; nt++)
      O[rowbase + nt * 16 + n16] = (__bf16)(acc_o[nt][r] * inv);
  }
}

// ---------------------------------------------------------------------------
// Kernel 3: out = attn @ Wproj + bproj (A bf16, output f32).
// M = 8192, N = 1024, K = 1024. Double-buffered LDS, 1 barrier / K-step.
// ---------------------------------------------------------------------------
__global__ __launch_bounds__(256) void proj_gemm_kernel(
    const __bf16* __restrict__ a_in, const float* __restrict__ w,
    const float* __restrict__ bias, float* __restrict__ out)
{
  constexpr int Kd = CD, N = CD, NSTEP = Kd / 32;
  __shared__ __align__(32) __bf16 sA[2 * 128 * GLDA];
  __shared__ __align__(32) __bf16 sB[2 * 64 * GLDB];

  const int tid  = threadIdx.x;
  const int m0   = blockIdx.x * 128;
  const int n0   = blockIdx.y * 64;
  const int wave = tid >> 5, lane = tid & 31;
  const int n16  = lane & 15, half = lane >> 4;
  const int wm0  = (wave & 3) * 32;
  const int wn0  = (wave >> 2) * 32;

  f32x8 acc[2][2] = {};

  g2s_a_bf16(a_in, sA, m0, 0, Kd, tid);
  g2s_b_f32(w, sB, n0, 0, N, tid);
  __syncthreads();

  for (int kt = 0; kt < NSTEP; kt++) {
    if (kt + 1 < NSTEP) {
      int nb = (kt + 1) & 1;
      g2s_a_bf16(a_in, sA + nb * 128 * GLDA, m0, (kt + 1) * 32, Kd, tid);
      g2s_b_f32(w, sB + nb * 64 * GLDB, n0, (kt + 1) * 32, N, tid);
    }
    mma_step(sA + (kt & 1) * 128 * GLDA, sB + (kt & 1) * 64 * GLDB,
             wm0, wn0, n16, half, acc);
    __syncthreads();
  }

  #pragma unroll
  for (int mt = 0; mt < 2; mt++) {
    #pragma unroll
    for (int nt = 0; nt < 2; nt++) {
      int gn = n0 + wn0 + nt * 16 + n16;
      float bs = bias[gn];
      #pragma unroll
      for (int r = 0; r < 8; r++) {
        int gm = m0 + wm0 + mt * 16 + 8 * half + r;
        out[(size_t)gm * N + gn] = acc[mt][nt][r] + bs;
      }
    }
  }
}

// ---------------------------------------------------------------------------
extern "C" void kernel_launch(void* const* d_in, const int* in_sizes, int n_in,
                              void* d_out, int out_size, void* d_ws, size_t ws_size,
                              hipStream_t stream) {
  (void)in_sizes; (void)n_in; (void)out_size; (void)ws_size;
  const float* x     = (const float*)d_in[0];
  const float* Wqkv  = (const float*)d_in[1];
  const float* bqkv  = (const float*)d_in[2];
  const float* Wproj = (const float*)d_in[3];
  const float* bproj = (const float*)d_in[4];
  float* out = (float*)d_out;

  const size_t elems = (size_t)CB * CH * CT * CHD;  // 8,388,608
  __bf16* wsq = (__bf16*)d_ws;
  __bf16* wsk = wsq + elems;
  __bf16* wsv = wsk + elems;   // V stored transposed [B,H,HD,T]
  __bf16* wsa = wsv + elems;

  dim3 blk(256);
  qkv_gemm_kernel<<<dim3(64, 48), blk, 0, stream>>>(x, Wqkv, bqkv, wsq, wsk, wsv);
  attn_fwd_kernel<<<dim3(16, 64), blk, 0, stream>>>(wsq, wsk, wsv, wsa);
  proj_gemm_kernel<<<dim3(64, 16), blk, 0, stream>>>(wsa, Wproj, bproj, out);
}
